// ShortDistanceAttention_88046829568180
// MI455X (gfx1250) — compile-verified, matrix operand order checked by reading, not run on hardware
//
#include <hip/hip_runtime.h>
#include <hip/hip_bf16.h>
#include <math.h>

#define N_NODES 8192
#define IN_F    256
#define OUT_F   256
#define ALPHA   0.2f

typedef __attribute__((ext_vector_type(2)))  float    v2f;
typedef __attribute__((ext_vector_type(8)))  float    v8f;
typedef __attribute__((ext_vector_type(16))) __bf16   v16bf;
typedef __attribute__((ext_vector_type(4)))  unsigned uvec4;

union BFrag { uvec4 u4[2]; v16bf bf; };

static __device__ __forceinline__ unsigned short f2bf(float f) {
  unsigned u = __float_as_uint(f);
  u += 0x7FFFu + ((u >> 16) & 1u);          // round-to-nearest-even
  return (unsigned short)(u >> 16);
}
static __device__ __forceinline__ unsigned pack2bf(float lo, float hi) {
  return (unsigned)f2bf(lo) | ((unsigned)f2bf(hi) << 16);
}

// ---------------------------------------------------------------------------
// K1: Wh = X @ W^T  (f32 WMMA 16x16x4).  4 waves/block, one 16x16 tile/wave.
// ---------------------------------------------------------------------------
__global__ void __launch_bounds__(128)
gemm_xwt_f32(const float* __restrict__ X, const float* __restrict__ W,
             float* __restrict__ Wh) {
  const int lane = threadIdx.x & 31;
  const int wid  = threadIdx.x >> 5;
  const int m0   = (blockIdx.x * 4 + wid) * 16;
  const int n0   = blockIdx.y * 16;
  const int half = lane >> 4;                 // 0: K+0/1, 1: K+2/3
  const int l16  = lane & 15;
  const float* ap = X + (size_t)(m0 + l16) * IN_F;
  const float* bp = W + (size_t)(n0 + l16) * IN_F;   // W^T tile == W rows
  v8f c = {};
  for (int k = 0; k < IN_F; k += 4) {
    const int ko = k + half * 2;
    v2f a = *(const v2f*)(ap + ko);
    v2f b = *(const v2f*)(bp + ko);
    c = __builtin_amdgcn_wmma_f32_16x16x4_f32(false, a, false, b,
                                              (short)0, c, false, false);
  }
  const int orow = m0 + half * 8;
  #pragma unroll
  for (int v = 0; v < 8; ++v)
    Wh[(size_t)(orow + v) * OUT_F + n0 + l16] = c[v];
}

// ---------------------------------------------------------------------------
// K2: s = Wh @ r_src, t = Wh @ r_dst, and WhbT[f][i] = bf16(Wh[i][f])
// ---------------------------------------------------------------------------
__global__ void __launch_bounds__(256)
project_rows(const float* __restrict__ Wh, const float* __restrict__ r,
             float* __restrict__ s, float* __restrict__ t,
             unsigned short* __restrict__ WhbT) {
  __shared__ float rs[256], rt[256];
  const int i   = blockIdx.x;
  const int tid = threadIdx.x;
  float w = Wh[(size_t)i * OUT_F + tid];
  WhbT[(size_t)tid * N_NODES + i] = f2bf(w);
  rs[tid] = w * r[tid];
  rt[tid] = w * r[OUT_F + tid];
  __syncthreads();
  for (int off = 128; off > 0; off >>= 1) {
    if (tid < off) { rs[tid] += rs[tid + off]; rt[tid] += rt[tid + off]; }
    __syncthreads();
  }
  if (tid == 0) { s[i] = rs[0]; t[i] = rt[0]; }
}

// ---------------------------------------------------------------------------
// K3a: per-row online softmax stats over masked leakyrelu(s_i + t_j)
//      (vectorized int4/float4 streaming of the A row)
// ---------------------------------------------------------------------------
__global__ void __launch_bounds__(256)
row_softmax_stats(const int* __restrict__ A, const float* __restrict__ s,
                  const float* __restrict__ t,
                  float* __restrict__ rowMax, float* __restrict__ rowInvZ) {
  __shared__ float sm[256], sz[256];
  const int i   = blockIdx.x;
  const int tid = threadIdx.x;
  const float si = s[i];
  const int* Arow = A + (size_t)i * N_NODES;
  float m = -INFINITY, z = 0.f;
  for (int j4 = tid * 4; j4 < N_NODES; j4 += 1024) {
    const int4   a  = *(const int4*)(Arow + j4);
    const float4 tv = *(const float4*)(t + j4);
    const int   av[4] = {a.x, a.y, a.z, a.w};
    const float tf[4] = {tv.x, tv.y, tv.z, tv.w};
    #pragma unroll
    for (int k = 0; k < 4; ++k) {
      if (av[k] != 0) {
        float e = si + tf[k];
        e = (e >= 0.f) ? e : ALPHA * e;
        float nm = fmaxf(m, e);
        z = z * __expf(m - nm) + __expf(e - nm);
        m = nm;
      }
    }
  }
  sm[tid] = m; sz[tid] = z;
  __syncthreads();
  for (int off = 128; off > 0; off >>= 1) {
    if (tid < off) {
      float m1 = sm[tid], m2 = sm[tid + off];
      float mm = fmaxf(m1, m2);
      float zz = (mm == -INFINITY) ? 0.f
               : sz[tid] * __expf(m1 - mm) + sz[tid + off] * __expf(m2 - mm);
      sm[tid] = mm; sz[tid] = zz;
    }
    __syncthreads();
  }
  if (tid == 0) {
    rowMax[i]  = sm[0];
    rowInvZ[i] = (sz[0] > 0.f) ? (1.f / sz[0]) : 0.f;
  }
}

// ---------------------------------------------------------------------------
// K3b: out = gelu( P @ Wh )   (bf16 WMMA 16x16x32, f32 accumulate)
//   block = 256 threads (8 waves), one 16-row M-tile, full 256 columns.
//   A chunks double-buffered into LDS with global_load_async_to_lds_b128
//   (ASYNCcnt); wave w owns rows 2w..2w+1 of the copy AND the P build, so
//   only s_wait_asynccnt (no extra barrier) orders producer->consumer.
//   One __syncthreads per 128-column chunk.
// ---------------------------------------------------------------------------
__global__ void __launch_bounds__(256)
attn_gemm_gelu(const int* __restrict__ A, const float* __restrict__ t,
               const float* __restrict__ s,
               const float* __restrict__ rowMax, const float* __restrict__ rowInvZ,
               const unsigned short* __restrict__ WhbT,
               float* __restrict__ out) {
  __shared__ int            Ab[2][16][128];        // 16 KB
  __shared__ unsigned short P[2][16][128];         //  8 KB
  __shared__ float s_si[16], s_m[16], s_iz[16];
  const int tid = threadIdx.x;
  const int m0  = blockIdx.x * 16;
  if (tid < 16) {
    s_si[tid] = s[m0 + tid];
    s_m[tid]  = rowMax[m0 + tid];
    s_iz[tid] = rowInvZ[m0 + tid];
  }

  const int lane = tid & 31;
  const int wid  = tid >> 5;
  const int l16  = lane & 15;
  const int half = lane >> 4;

  // per-wave row ownership for copy + P-build
  const int prow = 2 * wid + half;                 // 0..15
  const int pcol = l16 * 8;                        // 0..120, step 8

  // ---- async copy of one 16x128 int32 A chunk into Ab[buf] ----
  auto issue_async = [&](int jc, int buf) {
    const int* g = A + (size_t)(m0 + prow) * N_NODES + jc + pcol;
    unsigned ld  = (unsigned)(size_t)&Ab[buf][prow][pcol];
    asm volatile("global_load_async_to_lds_b128 %0, %1, off"
                 :: "v"(ld), "v"(g) : "memory");
    asm volatile("global_load_async_to_lds_b128 %0, %1, off offset:16"
                 :: "v"(ld), "v"(g) : "memory");
  };

  // ---- build normalized-attention bf16 rows (wave-local) ----
  auto compute_P = [&](int jc, int buf) {
    const int4 a0 = *(const int4*)&Ab[buf][prow][pcol];
    const int4 a1 = *(const int4*)&Ab[buf][prow][pcol + 4];
    const float4 t0 = *(const float4*)(t + jc + pcol);
    const float4 t1 = *(const float4*)(t + jc + pcol + 4);
    const float si = s_si[prow], mr = s_m[prow], iz = s_iz[prow];
    const int   av[8] = {a0.x, a0.y, a0.z, a0.w, a1.x, a1.y, a1.z, a1.w};
    const float tf[8] = {t0.x, t0.y, t0.z, t0.w, t1.x, t1.y, t1.z, t1.w};
    float pv[8];
    #pragma unroll
    for (int k = 0; k < 8; ++k) {
      float e = si + tf[k];
      e = (e >= 0.f) ? e : ALPHA * e;
      pv[k] = (av[k] != 0 && iz > 0.f) ? __expf(e - mr) * iz : 0.f;
    }
    uvec4 o = { pack2bf(pv[0], pv[1]), pack2bf(pv[2], pv[3]),
                pack2bf(pv[4], pv[5]), pack2bf(pv[6], pv[7]) };
    *(uvec4*)&P[buf][prow][pcol] = o;
  };

  const int n0a = wid * 32;
  const int n0b = n0a + 16;
  v8f c0 = {}, c1 = {};

  // prologue: stage chunk 0
  issue_async(0, 0);
  asm volatile("s_wait_asynccnt 0x0" ::: "memory");
  __syncthreads();                 // s_si/s_m/s_iz visible too
  compute_P(0, 0);

  for (int jc = 0; jc < N_NODES; jc += 128) {
    const int cur = (jc >> 7) & 1;
    const int nxt = cur ^ 1;
    const bool more = (jc + 128) < N_NODES;
    __syncthreads();               // P[cur] ready; P[nxt]/Ab[nxt] free
    if (more) issue_async(jc + 128, nxt);

    // 4 k-steps of K=32 bf16 WMMA for this 128-wide chunk
    #pragma unroll
    for (int kk = 0; kk < 4; ++kk) {
      const int kb = kk * 32;
      BFrag af, b0, b1;
      const unsigned short* ap = &P[cur][l16][kb + half * 8];
      af.u4[0] = *(const uvec4*)(ap);
      af.u4[1] = *(const uvec4*)(ap + 16);
      const unsigned short* bp0 =
          WhbT + (size_t)(n0a + l16) * N_NODES + jc + kb + half * 8;
      b0.u4[0] = *(const uvec4*)(bp0);
      b0.u4[1] = *(const uvec4*)(bp0 + 16);
      const unsigned short* bp1 =
          WhbT + (size_t)(n0b + l16) * N_NODES + jc + kb + half * 8;
      b1.u4[0] = *(const uvec4*)(bp1);
      b1.u4[1] = *(const uvec4*)(bp1 + 16);
      c0 = __builtin_amdgcn_wmma_f32_16x16x32_bf16(false, af.bf, false, b0.bf,
                                                   (short)0, c0, false, false);
      c1 = __builtin_amdgcn_wmma_f32_16x16x32_bf16(false, af.bf, false, b1.bf,
                                                   (short)0, c1, false, false);
    }

    if (more) {
      asm volatile("s_wait_asynccnt 0x0" ::: "memory");
      compute_P(jc + 128, nxt);    // wave-local rows: no extra barrier needed
    }
  }

  // exact GELU + store (C/D layout: row = v + 8*half, col = lane&15)
  const int orow = m0 + half * 8;
  #pragma unroll
  for (int v = 0; v < 8; ++v) {
    float x0 = c0[v], x1 = c1[v];
    float g0 = 0.5f * x0 * (1.f + erff(x0 * 0.70710678118654752f));
    float g1 = 0.5f * x1 * (1.f + erff(x1 * 0.70710678118654752f));
    out[(size_t)(orow + v) * OUT_F + n0a + l16] = g0;
    out[(size_t)(orow + v) * OUT_F + n0b + l16] = g1;
  }
}

// ---------------------------------------------------------------------------
extern "C" void kernel_launch(void* const* d_in, const int* in_sizes, int n_in,
                              void* d_out, int out_size, void* d_ws, size_t ws_size,
                              hipStream_t stream) {
  const float* X = (const float*)d_in[0];
  const int*   A = (const int*)  d_in[1];
  const float* W = (const float*)d_in[2];
  const float* r = (const float*)d_in[3];
  float* out = (float*)d_out;

  char* ws = (char*)d_ws;
  float*          Wh   = (float*)(ws);                                     // 8 MB
  unsigned short* WhbT = (unsigned short*)(ws + (size_t)8  * 1024 * 1024); // 4 MB
  float*          s    = (float*)(ws + (size_t)12 * 1024 * 1024);          // 32 KB
  float*          t    = (float*)(ws + (size_t)12 * 1024 * 1024 + 32 * 1024);
  float*          rmax = (float*)(ws + (size_t)12 * 1024 * 1024 + 64 * 1024);
  float*          riz  = (float*)(ws + (size_t)12 * 1024 * 1024 + 96 * 1024);

  gemm_xwt_f32<<<dim3(N_NODES / 64, OUT_F / 16), 128, 0, stream>>>(X, W, Wh);
  project_rows<<<N_NODES, 256, 0, stream>>>(Wh, r, s, t, WhbT);
  row_softmax_stats<<<N_NODES, 256, 0, stream>>>(A, s, t, rmax, riz);
  attn_gemm_gelu<<<N_NODES / 16, 256, 0, stream>>>(A, t, s, rmax, riz, WhbT, out);
}
